// FixedLiftCDEPdeNN6_51797305589796
// MI455X (gfx1250) — compile-verified, matrix operand order checked by reading, not run on hardware
//
#include <hip/hip_runtime.h>

// ---------------------------------------------------------------------------
// FixedLiftCDEPdeNN6 for MI455X (gfx1250, wave32).
// - Channel-mix GEMMs: v_wmma_f32_16x16x32_f16, A/B staged through LDS
//   (coalesced global loads, f16 conversion once, ds_load gathers).
// - Morph dilate/erode: LDS ring tiles; global->LDS fill uses the gfx1250
//   async-to-LDS path (ASYNCcnt) when the builtin is available.
// ---------------------------------------------------------------------------

typedef __attribute__((ext_vector_type(16))) _Float16 v16h;
typedef __attribute__((ext_vector_type(8)))  float    v8f;

#define BDIM   2
#define ODIM   8
#define HDIM   64
#define WDIM   64
#define HW     (HDIM * WDIM)        // 4096
#define OHW    (ODIM * HW)          // 32768
#define NEGV   (-1e30f)
#define BIGV   ( 1e30f)
#define NUH    2.1666667f           // NU/2 = (2*0.65/(2*0.65-1))/2

#if defined(__has_builtin)
#  if __has_builtin(__builtin_amdgcn_global_load_async_to_lds_b32)
#    define HAVE_ASYNC_LDS 1
#  endif
#endif
#ifndef HAVE_ASYNC_LDS
#  define HAVE_ASYNC_LDS 0
#endif

#if HAVE_ASYNC_LDS
typedef __attribute__((address_space(1))) int g_int;
typedef __attribute__((address_space(3))) int l_int;
#endif

// ---------------------------------------------------------------------------
// 1) Lift: 9x9 oriented cross-correlation, SAME zero padding.
// ---------------------------------------------------------------------------
__global__ void lift_kernel(const float* __restrict__ x,
                            const float* __restrict__ filt,
                            float* __restrict__ out) {
  int idx = blockIdx.x * blockDim.x + threadIdx.x;   // B*3*O*HW = 196608
  if (idx >= BDIM * 3 * OHW) return;
  int w  = idx & 63;
  int h  = (idx >> 6) & 63;
  int o  = (idx >> 12) & 7;
  int bc = idx >> 15;                                // b*3 + c
  const float* xi = x + (size_t)bc * HW;
  const float* f  = filt + o * 81;
  float acc = 0.f;
  #pragma unroll
  for (int i = 0; i < 9; ++i) {
    int hh = h + i - 4;
    if ((unsigned)hh >= (unsigned)HDIM) continue;
    #pragma unroll
    for (int j = 0; j < 9; ++j) {
      int ww = w + j - 4;
      if ((unsigned)ww >= (unsigned)WDIM) continue;
      acc += f[i * 9 + j] * xi[hh * WDIM + ww];
    }
  }
  out[idx] = acc;
}

// ---------------------------------------------------------------------------
// 2) Small-K channel mix (VALU; used for lin0 where K=3).
// ---------------------------------------------------------------------------
__global__ void linmix_kernel(const float* __restrict__ in,
                              const float* __restrict__ w,
                              float* __restrict__ out, int Cin, int Cout) {
  int idx = blockIdx.x * blockDim.x + threadIdx.x;
  int total = BDIM * Cout * OHW;
  if (idx >= total) return;
  int p = idx & (OHW - 1);
  int d = (idx >> 15) % Cout;
  int b = idx / (Cout * OHW);
  const float* xi = in + (size_t)b * Cin * OHW + p;
  const float* wd = w + d * Cin;
  float acc = 0.f;
  for (int c = 0; c < Cin; ++c) acc += wd[c] * xi[(size_t)c * OHW];
  out[idx] = acc;
}

// ---------------------------------------------------------------------------
// 3) Convection on M2: orientation roll fused with rotated spatial shift.
// ---------------------------------------------------------------------------
__global__ void convect_kernel(const float* __restrict__ in,
                               const float* __restrict__ vec,
                               float* __restrict__ out, int C) {
  int idx = blockIdx.x * blockDim.x + threadIdx.x;
  if (idx >= BDIM * C * OHW) return;
  int w = idx & 63;
  int h = (idx >> 6) & 63;
  int o = (idx >> 12) & 7;
  int c = (idx >> 15) % C;
  int b = idx / (C * OHW);

  float cth = vec[c * 3 + 0];
  float cx  = vec[c * 3 + 1];
  float cy  = vec[c * 3 + 2];

  float pos = (float)o - cth;
  float i0f = floorf(pos);
  float wf  = pos - i0f;
  int i0 = (((int)i0f % 8) + 8) & 7;
  int i1 = (i0 + 1) & 7;

  float th = 0.78539816339744831f * (float)o;        // 2*pi*o/8
  float st, ct;
  __sincosf(th, &st, &ct);
  float sx = cx * ct - cy * st;
  float sy = cx * st + cy * ct;

  float Y = (float)h - sy;
  float X = (float)w - sx;
  float y0f = floorf(Y), x0f = floorf(X);
  float ty = Y - y0f, tx = X - x0f;
  int y0 = (int)y0f, x0 = (int)x0f;
  int iy0 = min(max(y0, 0), HDIM - 1);
  int iy1 = min(max(y0 + 1, 0), HDIM - 1);
  int ix0 = min(max(x0, 0), WDIM - 1);
  int ix1 = min(max(x0 + 1, 0), WDIM - 1);

  const float* g0 = in + ((size_t)(b * C + c) * ODIM + i0) * HW;
  const float* g1 = in + ((size_t)(b * C + c) * ODIM + i1) * HW;
  float w00 = (1.f - ty) * (1.f - tx), w01 = (1.f - ty) * tx;
  float w10 = ty * (1.f - tx),         w11 = ty * tx;
  float v0 = w00 * g0[iy0 * WDIM + ix0] + w01 * g0[iy0 * WDIM + ix1]
           + w10 * g0[iy1 * WDIM + ix0] + w11 * g0[iy1 * WDIM + ix1];
  float v1 = w00 * g1[iy0 * WDIM + ix0] + w01 * g1[iy0 * WDIM + ix1]
           + w10 * g1[iy1 * WDIM + ix0] + w11 * g1[iy1 * WDIM + ix1];
  out[idx] = (1.f - wf) * v0 + wf * v1;
}

// ---------------------------------------------------------------------------
// 4) Batchnorm stats + apply.
// ---------------------------------------------------------------------------
__global__ void bn_stats_kernel(const float* __restrict__ x,
                                float* __restrict__ stats, int C) {
  int c = blockIdx.x;
  __shared__ float ssum[256];
  __shared__ float ssq[256];
  float s = 0.f, q = 0.f;
  for (int i = threadIdx.x; i < BDIM * OHW; i += 256) {
    int b = i >> 15;
    int p = i & (OHW - 1);
    float v = x[(size_t)(b * C + c) * OHW + p];
    s += v;
    q += v * v;
  }
  ssum[threadIdx.x] = s;
  ssq[threadIdx.x]  = q;
  __syncthreads();
  for (int off = 128; off > 0; off >>= 1) {
    if ((int)threadIdx.x < off) {
      ssum[threadIdx.x] += ssum[threadIdx.x + off];
      ssq[threadIdx.x]  += ssq[threadIdx.x + off];
    }
    __syncthreads();
  }
  if (threadIdx.x == 0) {
    float inv = 1.f / (float)(BDIM * OHW);
    float mu  = ssum[0] * inv;
    float var = ssq[0] * inv - mu * mu;
    stats[2 * c]     = mu;
    stats[2 * c + 1] = var;
  }
}

__global__ void bn_apply_kernel(float* __restrict__ x,
                                const float* __restrict__ stats,
                                const float* __restrict__ g,
                                const float* __restrict__ bta,
                                int C, int relu) {
  int idx = blockIdx.x * blockDim.x + threadIdx.x;
  if (idx >= BDIM * C * OHW) return;
  int c = (idx >> 15) % C;
  float mu  = stats[2 * c];
  float var = stats[2 * c + 1];
  float v = g[c] * (x[idx] - mu) * rsqrtf(var + 1e-5f) + bta[c];
  if (relu) v = fmaxf(v, 0.f);
  x[idx] = v;
}

// ---------------------------------------------------------------------------
// 5) Morphological structuring function: (softplus(g)+0.1) . d^2, ^(NU/2)
// ---------------------------------------------------------------------------
__global__ void morphk_kernel(const float* __restrict__ g,
                              float* __restrict__ kern, int C) {
  int idx = blockIdx.x * blockDim.x + threadIdx.x;
  if (idx >= C * 125) return;
  int c = idx / 125;
  int t = idx % 125;
  float dz = (float)(t / 25 - 2);
  float dy = (float)((t / 5) % 5 - 2);
  float dx = (float)(t % 5 - 2);
  auto sp = [](float v) {
    return (v > 0.f ? v + log1pf(__expf(-v)) : log1pf(__expf(v))) + 0.1f;
  };
  float d2 = sp(g[c * 3 + 0]) * dz * dz
           + sp(g[c * 3 + 1]) * dy * dy
           + sp(g[c * 3 + 2]) * dx * dx;
  kern[idx] = powf(d2, NUH);
}

// ---------------------------------------------------------------------------
// 6) Dilate (ER=0: max(f - k), OOB=-inf) / Erode (ER=1: min(f + k), OOB=+inf).
//    LDS tile: orientation ring (8) x 20x20 halo. Tile fill uses gfx1250
//    async global->LDS copies when available (ASYNCcnt), PADV halo via DS
//    stores at disjoint addresses, s_wait_asynccnt before the barrier.
// ---------------------------------------------------------------------------
template <int ER>
__global__ void morph_de_kernel(const float* __restrict__ in,
                                const float* __restrict__ kern,
                                float* __restrict__ out, int C) {
  __shared__ float tile[ODIM * 20 * 20];   // 12.8 KB
  __shared__ float kk[125];
  int tileid = blockIdx.x;                 // 16 tiles of 16x16
  int c = blockIdx.y;
  int b = blockIdx.z;
  int h0 = (tileid >> 2) * 16;
  int w0 = (tileid & 3) * 16;
  const float* base = in + (size_t)(b * C + c) * OHW;
  const float PADV = ER ? BIGV : NEGV;

  for (int t = threadIdx.x; t < ODIM * 400; t += 256) {
    int o = t / 400;
    int r = (t % 400) / 20;
    int q = t % 20;
    int gh = h0 + r - 2, gw = w0 + q - 2;
    if ((unsigned)gh < (unsigned)HDIM && (unsigned)gw < (unsigned)WDIM) {
#if HAVE_ASYNC_LDS
      __builtin_amdgcn_global_load_async_to_lds_b32(
          (g_int*)(base + o * HW + gh * WDIM + gw),
          (l_int*)(&tile[t]),
          0, 0);
#else
      tile[t] = base[o * HW + gh * WDIM + gw];
#endif
    } else {
      tile[t] = PADV;
    }
  }
  for (int t = threadIdx.x; t < 125; t += 256) kk[t] = kern[c * 125 + t];
#if HAVE_ASYNC_LDS
  asm volatile("s_wait_asynccnt 0" ::: "memory");
#endif
  __syncthreads();

  int r = threadIdx.x >> 4;
  int q = threadIdx.x & 15;
  float acc[ODIM];
  #pragma unroll
  for (int o = 0; o < ODIM; ++o) acc[o] = PADV;

  for (int dz = 0; dz < 5; ++dz) {
    for (int dy = 0; dy < 5; ++dy) {
      #pragma unroll
      for (int dx = 0; dx < 5; ++dx) {
        float kv = kk[(dz * 5 + dy) * 5 + dx];
        int sb = (r + dy) * 20 + (q + dx);
        #pragma unroll
        for (int o = 0; o < ODIM; ++o) {
          float v = tile[((o + dz - 2) & 7) * 400 + sb];
          if (ER) acc[o] = fminf(acc[o], v + kv);
          else    acc[o] = fmaxf(acc[o], v - kv);
        }
      }
    }
  }
  float* ob = out + (size_t)(b * C + c) * OHW + h0 * WDIM + w0;
  #pragma unroll
  for (int o = 0; o < ODIM; ++o) ob[o * HW + r * WDIM + q] = acc[o];
}

// ---------------------------------------------------------------------------
// 7) Channel mix via WMMA, LDS-staged.
//    out(B,16,OHW) = in(B,Cin,OHW) x w(16,Cin)^T; K = Cin zero-padded to 32.
//    Block: 256 threads = 8 waves = 8 M-tiles (128 consecutive positions).
//    Activations panel [Cin][128] and weights [16][32] staged in LDS as f16
//    via coalesced global loads; fragments gathered with ds loads.
// ---------------------------------------------------------------------------
__global__ void linmix_wmma_kernel(const float* __restrict__ in,
                                   const float* __restrict__ w,
                                   float* __restrict__ out, int Cin) {
  __shared__ _Float16 sh[32 * 128];   // [k][p_local], 8 KB
  __shared__ _Float16 shw[16 * 32];   // [n][k], 1 KB
  int tid = threadIdx.x;
  int b    = blockIdx.x >> 8;                 // 512 blocks: 2 x 256 chunks
  int pblk = (blockIdx.x & 255) * 128;

  // Stage activations panel (coalesced along positions).
  const float* xb = in + (size_t)b * Cin * OHW + pblk;
  for (int idx = tid; idx < Cin * 128; idx += 256) {
    int k  = idx >> 7;
    int pl = idx & 127;
    sh[idx] = (_Float16)xb[(size_t)k * OHW + pl];
  }
  // Stage weights, zero-padded to K=32.
  for (int idx = tid; idx < 16 * 32; idx += 256) {
    int n = idx >> 5;
    int k = idx & 31;
    shw[idx] = (k < Cin) ? (_Float16)w[n * Cin + k] : (_Float16)0.f;
  }
  __syncthreads();

  int wave = tid >> 5;
  int lane = tid & 31;
  int row   = lane & 15;                      // A row (M) / B col (N)
  int khalf = lane >> 4;                      // K half selector
  int pl0   = wave * 16;                      // tile's first local position

  v16h a{}, bf{};
  #pragma unroll
  for (int i = 0; i < 16; ++i) {
    // lane<16: K in {0..7, 16..23};  lane>=16: K in {8..15, 24..31}
    int k = ((i & 8) << 1) + khalf * 8 + (i & 7);
    a[i]  = (k < Cin) ? sh[k * 128 + pl0 + row] : (_Float16)0.f;
    bf[i] = shw[row * 32 + k];
  }
  v8f cc = {};
  cc = __builtin_amdgcn_wmma_f32_16x16x32_f16(
      /*neg_a=*/false, a, /*neg_b=*/false, bf,
      /*c_mod=*/(short)0, cc, /*reuse_a=*/false, /*reuse_b=*/false);

  // D: lane<16 -> M=v, lane>=16 -> M=8+v; N=lane&15. Contiguous in M.
  float* ob = out + (size_t)b * 16 * OHW + (size_t)row * OHW
            + pblk + pl0 + khalf * 8;
  #pragma unroll
  for (int v = 0; v < 8; ++v) ob[v] = cc[v];
}

// ---------------------------------------------------------------------------
// 8) Head: max over orientation, 1-channel linear, sigmoid.
// ---------------------------------------------------------------------------
__global__ void head_kernel(const float* __restrict__ z,
                            const float* __restrict__ fw,
                            float* __restrict__ out) {
  int idx = blockIdx.x * blockDim.x + threadIdx.x;   // B*HW = 8192
  if (idx >= BDIM * HW) return;
  int p = idx & (HW - 1);
  int b = idx >> 12;
  float acc = 0.f;
  for (int c = 0; c < 16; ++c) {
    const float* zc = z + (size_t)(b * 16 + c) * OHW + p;
    float m = NEGV;
    #pragma unroll
    for (int o = 0; o < ODIM; ++o) m = fmaxf(m, zc[o * HW]);
    acc += fw[c] * m;
  }
  out[idx] = 1.f / (1.f + __expf(-acc));
}

// ---------------------------------------------------------------------------
// Orchestration
// ---------------------------------------------------------------------------
extern "C" void kernel_launch(void* const* d_in, const int* in_sizes, int n_in,
                              void* d_out, int out_size, void* d_ws, size_t ws_size,
                              hipStream_t stream) {
  (void)in_sizes; (void)n_in; (void)out_size; (void)ws_size;

  const float* x     = (const float*)d_in[0];
  const float* liftf = (const float*)d_in[1];
  const float* lin0w = (const float*)d_in[2];
  const float* vec0  = (const float*)d_in[3];
  const float* bn0g  = (const float*)d_in[4];
  const float* bn0b  = (const float*)d_in[5];
  const float* finw  = (const float*)d_in[30];

  float* ws    = (float*)d_ws;
  float* z0    = ws;                   // 393216  (B,3,8,HW)
  float* bufA  = z0 + 393216;          // 2097152 (B,32,8,HW)
  float* bufB  = bufA + 2097152;       // 2097152
  float* kernD = bufB + 2097152;       // 4000
  float* kernE = kernD + 4096;         // 4000
  float* stats = kernE + 4096;         // 64

  // lift + initial channel mix (K=3 -> VALU)
  lift_kernel<<<(BDIM * 3 * OHW + 255) / 256, 256, 0, stream>>>(x, liftf, z0);
  linmix_kernel<<<(BDIM * 32 * OHW + 255) / 256, 256, 0, stream>>>(z0, lin0w, bufA, 3, 32);
  convect_kernel<<<(BDIM * 32 * OHW + 255) / 256, 256, 0, stream>>>(bufA, vec0, bufB, 32);
  bn_stats_kernel<<<32, 256, 0, stream>>>(bufB, stats, 32);
  bn_apply_kernel<<<(BDIM * 32 * OHW + 255) / 256, 256, 0, stream>>>(bufB, stats, bn0g, bn0b, 32, 0);

  float* cur = bufB;
  float* oth = bufA;
  const int cins[4] = {32, 16, 16, 16};

  for (int L = 0; L < 4; ++L) {
    const float* lvec = (const float*)d_in[6 + L * 6 + 0];
    const float* lgd  = (const float*)d_in[6 + L * 6 + 1];
    const float* lge  = (const float*)d_in[6 + L * 6 + 2];
    const float* lw   = (const float*)d_in[6 + L * 6 + 3];
    const float* lg   = (const float*)d_in[6 + L * 6 + 4];
    const float* lb   = (const float*)d_in[6 + L * 6 + 5];
    int C = cins[L];
    int nElem = BDIM * C * OHW;

    convect_kernel<<<(nElem + 255) / 256, 256, 0, stream>>>(cur, lvec, oth, C);
    morphk_kernel<<<(C * 125 + 255) / 256, 256, 0, stream>>>(lgd, kernD, C);
    morphk_kernel<<<(C * 125 + 255) / 256, 256, 0, stream>>>(lge, kernE, C);
    morph_de_kernel<0><<<dim3(16, C, BDIM), 256, 0, stream>>>(oth, kernD, cur, C);
    morph_de_kernel<1><<<dim3(16, C, BDIM), 256, 0, stream>>>(cur, kernE, oth, C);
    linmix_wmma_kernel<<<512, 256, 0, stream>>>(oth, lw, cur, C);
    bn_stats_kernel<<<16, 256, 0, stream>>>(cur, stats, 16);
    bn_apply_kernel<<<(BDIM * 16 * OHW + 255) / 256, 256, 0, stream>>>(cur, stats, lg, lb, 16, 1);
  }

  head_kernel<<<(BDIM * HW + 255) / 256, 256, 0, stream>>>(cur, finw, (float*)d_out);
}
